// GraphTransformerBaseMapper_386547057156
// MI455X (gfx1250) — matrix operand order repaired and checked by reading, not compiled, not run on hardware
//
#include <hip/hip_runtime.h>
#include <hip/hip_bf16.h>
#include <math.h>

typedef __attribute__((ext_vector_type(16))) __bf16       v16bf;
typedef __attribute__((ext_vector_type(8)))  float        v8f;
typedef __attribute__((ext_vector_type(4)))  unsigned int u32x4;

#define HIDC   512
#define MLPC   2048
#define HEADSC 16
#define HDC    32
#define SRCG   40320
#define DSTG   10240
#define E0C    81920

// ---------------------------------------------------------------------------
// order-preserving float<->uint encoding for atomic max
// ---------------------------------------------------------------------------
__device__ __forceinline__ unsigned encF(float f) {
  unsigned u = __float_as_uint(f);
  return (u & 0x80000000u) ? ~u : (u | 0x80000000u);
}
__device__ __forceinline__ float decF(unsigned u) {
  u = (u & 0x80000000u) ? (u & 0x7FFFFFFFu) : ~u;
  return __uint_as_float(u);
}
#define ENC_NEG_INF 0x007FFFFFu   // encF(-inf)

// ---------------------------------------------------------------------------
// generic fill
// ---------------------------------------------------------------------------
__global__ void gt_fill_u32(unsigned* __restrict__ p, unsigned v, size_t n) {
  size_t i = (size_t)blockIdx.x * blockDim.x + threadIdx.x;
  if (i < n) p[i] = v;
}

// fp32 [K,N] -> bf16 transposed [N,K]  (weights; one pass per launch)
__global__ __launch_bounds__(256) void gt_cvt_bf16_t(
    const float* __restrict__ in, __bf16* __restrict__ out, int K, int N) {
  __shared__ float t[32][33];
  int n0 = blockIdx.x * 32, k0 = blockIdx.y * 32;
  int tx = threadIdx.x, ty = threadIdx.y;
#pragma unroll
  for (int i = 0; i < 32; i += 8)
    t[ty + i][tx] = in[(size_t)(k0 + ty + i) * N + n0 + tx];
  __syncthreads();
#pragma unroll
  for (int i = 0; i < 32; i += 8)
    out[(size_t)(n0 + ty + i) * K + k0 + tx] = (__bf16)t[tx][ty + i];
}

// (a + b) -> bf16
__global__ void gt_add_cvt_bf16(const float* __restrict__ a, const float* __restrict__ b,
                                __bf16* __restrict__ out, size_t n) {
  size_t i = (size_t)blockIdx.x * blockDim.x + threadIdx.x;
  if (i < n) out[i] = (__bf16)(a[i] + b[i]);
}

// ---------------------------------------------------------------------------
// LayerNorm over 512 features, emit bf16 (feeds WMMA GEMMs)
// ---------------------------------------------------------------------------
__global__ __launch_bounds__(128) void gt_ln_bf16(
    const float* __restrict__ x, const float* __restrict__ g, const float* __restrict__ b,
    __bf16* __restrict__ out, int rows) {
  int row = blockIdx.x;
  if (row >= rows) return;
  const float* xr = x + (size_t)row * HIDC;
  float v[4], s = 0.f, s2 = 0.f;
#pragma unroll
  for (int i = 0; i < 4; i++) {
    v[i] = xr[threadIdx.x + i * 128];
    s += v[i]; s2 += v[i] * v[i];
  }
  __shared__ float r1[128], r2[128];
  r1[threadIdx.x] = s; r2[threadIdx.x] = s2;
  __syncthreads();
  for (int st = 64; st > 0; st >>= 1) {
    if ((int)threadIdx.x < st) {
      r1[threadIdx.x] += r1[threadIdx.x + st];
      r2[threadIdx.x] += r2[threadIdx.x + st];
    }
    __syncthreads();
  }
  float mean = r1[0] * (1.f / HIDC);
  float var  = r2[0] * (1.f / HIDC) - mean * mean;
  float rstd = rsqrtf(var + 1e-5f);
#pragma unroll
  for (int i = 0; i < 4; i++) {
    int c = threadIdx.x + i * 128;
    out[(size_t)row * HIDC + c] = (__bf16)((v[i] - mean) * rstd * g[c] + b[c]);
  }
}

// ---------------------------------------------------------------------------
// bf16 WMMA GEMM: C[M,N] = A[M,K] @ Bt[N,K]^T (+bias) (+gelu) (+resid)
// - B pre-transposed, so both tiles stage as contiguous b128 async copies
// - double-buffered LDS: tile t+1 streams in (ASYNCcnt) while tile t computes
// 256 threads = 8 waves (wave32); block tile 128x128; K-step 32
// wave grid 4(M) x 2(N); each wave: 32x64 via 2x4 v_wmma_f32_16x16x32_bf16
// M,N multiples of 128; K multiple of 32 (true for all call sites)
// ---------------------------------------------------------------------------
__global__ __launch_bounds__(256) void gt_gemm_bf16(
    const __bf16* __restrict__ A, const __bf16* __restrict__ Bt,
    const float* __restrict__ bias, const float* __restrict__ resid,
    float* __restrict__ Cf, __bf16* __restrict__ Cb,
    int M, int N, int K, int gelu) {
  __shared__ __bf16 lA[2][128 * 32];   // [m][k] row-major, double buffered
  __shared__ __bf16 lB[2][128 * 32];   // [n][k] row-major (B pre-transposed)
  const int tid  = threadIdx.x;
  const int lane = tid & 31;
  const int wave = tid >> 5;
  const int m0 = blockIdx.y * 128, n0 = blockIdx.x * 128;
  const int wm = (wave & 3) * 32;   // wave M offset
  const int wn = (wave >> 2) * 64;  // wave N offset

  v8f acc[2][4];
#pragma unroll
  for (int a = 0; a < 2; a++)
#pragma unroll
    for (int c = 0; c < 4; c++)
#pragma unroll
      for (int r = 0; r < 8; r++) acc[a][c][r] = 0.f;

  const int rowT = tid >> 1, colT = (tid & 1) * 16;   // 128 rows x 32 bf16 per tile
  const __bf16* gA = A  + (size_t)(m0 + rowT) * K + colT;
  const __bf16* gB = Bt + (size_t)(n0 + rowT) * K + colT;
  unsigned laOff[2], lbOff[2];
#pragma unroll
  for (int bfi = 0; bfi < 2; bfi++) {
    laOff[bfi] = (unsigned)(uintptr_t)&lA[bfi][rowT * 32 + colT];
    lbOff[bfi] = (unsigned)(uintptr_t)&lB[bfi][rowT * 32 + colT];
  }

  // async global->LDS copy of one 128x32 A tile + 128x32 B tile into buffer `bfi`
  auto issue = [&](int bfi, int k0) {
    asm volatile("global_load_async_to_lds_b128 %0, %1, off"
                 :: "v"(laOff[bfi]), "v"(gA + k0) : "memory");
    asm volatile("global_load_async_to_lds_b128 %0, %1, off offset:16"
                 :: "v"(laOff[bfi]), "v"(gA + k0) : "memory");
    asm volatile("global_load_async_to_lds_b128 %0, %1, off"
                 :: "v"(lbOff[bfi]), "v"(gB + k0) : "memory");
    asm volatile("global_load_async_to_lds_b128 %0, %1, off offset:16"
                 :: "v"(lbOff[bfi]), "v"(gB + k0) : "memory");
  };

  const int nsteps = K >> 5;
  issue(0, 0);
  for (int t = 0; t < nsteps; ++t) {
    const int cur = t & 1;
    asm volatile("s_wait_asynccnt 0" ::: "memory");
    __syncthreads();                      // tile t visible to all waves
    if (t + 1 < nsteps) issue(cur ^ 1, (t + 1) << 5);  // overlap next copy with compute

    // A frag (16x32 bf16): lanes 0-15 rows M=lane, K {0..7,16..23}; lanes 16-31 K {8..15,24..31}
    union { u32x4 q[2]; v16bf h; } fa[2], fb[4];
    const int r  = lane & 15;
    const int kb = (lane >> 4) * 8;
#pragma unroll
    for (int mi = 0; mi < 2; mi++) {
      const __bf16* base = &lA[cur][(wm + mi * 16 + r) * 32];
      fa[mi].q[0] = *(const u32x4*)(base + kb);
      fa[mi].q[1] = *(const u32x4*)(base + 16 + kb);
    }
    // B frag (32x16 bf16): lanes 0-15 col N=lane, K {0..7,16..23}; lanes 16-31 K {8..15,24..31}
    const int c  = lane & 15;
    const int ko = (lane >> 4) * 8;
#pragma unroll
    for (int ni = 0; ni < 4; ni++) {
      const __bf16* base = &lB[cur][(wn + ni * 16 + c) * 32 + ko];
      fb[ni].q[0] = *(const u32x4*)(base);
      fb[ni].q[1] = *(const u32x4*)(base + 16);
    }
#pragma unroll
    for (int mi = 0; mi < 2; mi++)
#pragma unroll
      for (int ni = 0; ni < 4; ni++)
        acc[mi][ni] = __builtin_amdgcn_wmma_f32_16x16x32_bf16(
            false, fa[mi].h, false, fb[ni].h, (short)0, acc[mi][ni], false, false);
    __syncthreads();                      // reads of buffer `cur` done before it is refilled
  }

  // epilogue: C/D layout -> VGPR r holds M = r + (lane>=16 ? 8 : 0), N = lane&15
  const int cN   = lane & 15;
  const int rOff = (lane >> 4) * 8;
#pragma unroll
  for (int mi = 0; mi < 2; mi++)
#pragma unroll
    for (int ni = 0; ni < 4; ni++) {
      int colg = n0 + wn + ni * 16 + cN;
      float bvv = bias ? bias[colg] : 0.f;
#pragma unroll
      for (int r = 0; r < 8; r++) {
        int rowg = m0 + wm + mi * 16 + rOff + r;
        float x = acc[mi][ni][r] + bvv;
        if (gelu) x = 0.5f * x * (1.f + erff(x * 0.70710678118654752f));
        size_t idx = (size_t)rowg * N + colg;
        if (resid) x += resid[idx];
        if (Cf) Cf[idx] = x;
        if (Cb) Cb[idx] = (__bf16)x;
      }
    }
}

// ---------------------------------------------------------------------------
// edge embedding: e = concat(edge_attr_base, trainable)[e % E0] @ We + be  (K=11)
// ---------------------------------------------------------------------------
__global__ __launch_bounds__(128) void gt_edge_embed(
    const float* __restrict__ eab, const float* __restrict__ trn,
    const float* __restrict__ We, const float* __restrict__ be,
    __bf16* __restrict__ eout, int E) {
  int e = blockIdx.x;
  if (e >= E) return;
  int eb = e % E0C;
  __shared__ float f[11];
  if (threadIdx.x < 3)       f[threadIdx.x] = eab[(size_t)eb * 3 + threadIdx.x];
  else if (threadIdx.x < 11) f[threadIdx.x] = trn[(size_t)eb * 8 + threadIdx.x - 3];
  __syncthreads();
#pragma unroll
  for (int c4 = 0; c4 < 4; c4++) {
    int col = threadIdx.x + c4 * 128;
    float acc = be[col];
#pragma unroll
    for (int j = 0; j < 11; j++) acc += f[j] * We[j * HIDC + col];
    eout[(size_t)e * HIDC + col] = (__bf16)acc;
  }
}

// ---------------------------------------------------------------------------
// attention pass 1: alpha = dot(q[dst], k[src]+e)/sqrt(HD); atomic max per (dst,h)
// ---------------------------------------------------------------------------
__global__ __launch_bounds__(256) void gt_alpha(
    const float* __restrict__ q, const __bf16* __restrict__ k, const __bf16* __restrict__ e,
    const int* __restrict__ rs, const int* __restrict__ rd,
    float* __restrict__ alpha, unsigned* __restrict__ amax, int E) {
  int idx = blockIdx.x * 256 + threadIdx.x;
  if (idx >= E * HEADSC) return;
  int h = idx & 15, ed = idx >> 4;
  int bi = ed / E0C, eb = ed - bi * E0C;
  int src = rs[eb] + bi * SRCG;
  int dst = rd[eb] + bi * DSTG;
  const float*  qp = q + (size_t)dst * HIDC + h * HDC;
  const __bf16* kp = k + (size_t)src * HIDC + h * HDC;
  const __bf16* ep = e + (size_t)ed  * HIDC + h * HDC;
  float s = 0.f;
#pragma unroll
  for (int i = 0; i < HDC; i++) s += qp[i] * ((float)kp[i] + (float)ep[i]);
  s *= 0.17677669529663687f;  // 1/sqrt(32)
  alpha[idx] = s;
  atomicMax(&amax[dst * HEADSC + h], encF(s));
}

// pass 2: ex = exp(alpha - amax[dst]); atomicAdd denom (alpha overwritten in place)
__global__ __launch_bounds__(256) void gt_expsum(
    const unsigned* __restrict__ amax, const int* __restrict__ rd,
    float* __restrict__ alpha, float* __restrict__ denom, int E) {
  int idx = blockIdx.x * 256 + threadIdx.x;
  if (idx >= E * HEADSC) return;
  int h = idx & 15, ed = idx >> 4;
  int bi = ed / E0C, eb = ed - bi * E0C;
  int dst = rd[eb] + bi * DSTG;
  float m = decF(amax[dst * HEADSC + h]);
  if (!isfinite(m)) m = 0.f;
  float ex = expf(alpha[idx] - m);
  alpha[idx] = ex;
  atomicAdd(&denom[dst * HEADSC + h], ex);
}

// pass 3: out_acc[dst] += (v[src]+e) * ex/(denom+eps)
__global__ __launch_bounds__(256) void gt_msg(
    const __bf16* __restrict__ v, const __bf16* __restrict__ e,
    const float* __restrict__ ex, const float* __restrict__ denom,
    const int* __restrict__ rs, const int* __restrict__ rd,
    float* __restrict__ outacc, int E) {
  int idx = blockIdx.x * 256 + threadIdx.x;
  if (idx >= E * HEADSC) return;
  int h = idx & 15, ed = idx >> 4;
  int bi = ed / E0C, eb = ed - bi * E0C;
  int src = rs[eb] + bi * SRCG;
  int dst = rd[eb] + bi * DSTG;
  float w = ex[idx] / (denom[dst * HEADSC + h] + 1e-16f);
  const __bf16* vp = v + (size_t)src * HIDC + h * HDC;
  const __bf16* ep = e + (size_t)ed  * HIDC + h * HDC;
  float* op = outacc + (size_t)dst * HIDC + h * HDC;
#pragma unroll 4
  for (int i = 0; i < HDC; i++)
    atomicAdd(op + i, ((float)vp[i] + (float)ep[i]) * w);
}

// ---------------------------------------------------------------------------
extern "C" void kernel_launch(void* const* d_in, const int* in_sizes, int n_in,
                              void* d_out, int out_size, void* d_ws, size_t ws_size,
                              hipStream_t stream) {
  const float* x_src = (const float*)d_in[0];
  const float* x_dst = (const float*)d_in[1];
  const float* eab   = (const float*)d_in[2];
  const float* trn   = (const float*)d_in[3];
  const float* Wq = (const float*)d_in[4];  const float* bq = (const float*)d_in[5];
  const float* Wk = (const float*)d_in[6];  const float* bk = (const float*)d_in[7];
  const float* Wv = (const float*)d_in[8];  const float* bv = (const float*)d_in[9];
  const float* Ws = (const float*)d_in[10]; const float* bs = (const float*)d_in[11];
  const float* We = (const float*)d_in[12]; const float* be = (const float*)d_in[13];
  const float* Wp = (const float*)d_in[14]; const float* bp = (const float*)d_in[15];
  const float* ln1g = (const float*)d_in[16]; const float* ln1b = (const float*)d_in[17];
  const float* ln2g = (const float*)d_in[18]; const float* ln2b = (const float*)d_in[19];
  const float* mlng = (const float*)d_in[20]; const float* mlnb = (const float*)d_in[21];
  const float* W1 = (const float*)d_in[22]; const float* b1 = (const float*)d_in[23];
  const float* W2 = (const float*)d_in[24]; const float* b2 = (const float*)d_in[25];
  const int* eib = (const int*)d_in[26];
  (void)n_in; (void)out_size; (void)ws_size;

  const int B  = in_sizes[0] / (SRCG * HIDC);
  const int Ms = B * SRCG;     // src rows (multiple of 128)
  const int Md = B * DSTG;     // dst rows (multiple of 128)
  const int E  = B * E0C;      // edges
  const int* row_s = eib;
  const int* row_d = eib + E0C;

  // ---- workspace bump allocator (256B aligned) ----
  char* base = (char*)d_ws; size_t off = 0;
  auto alloc = [&](size_t bytes) -> void* {
    void* r = base + off;
    off = (off + bytes + 255) & ~(size_t)255;
    return r;
  };
  __bf16* hs   = (__bf16*)alloc((size_t)Ms * HIDC * 2);
  __bf16* hd   = (__bf16*)alloc((size_t)Md * HIDC * 2);
  __bf16* wqb  = (__bf16*)alloc((size_t)HIDC * HIDC * 2);   // all weight copies transposed [N,K]
  __bf16* wkb  = (__bf16*)alloc((size_t)HIDC * HIDC * 2);
  __bf16* wvb  = (__bf16*)alloc((size_t)HIDC * HIDC * 2);
  __bf16* wsb  = (__bf16*)alloc((size_t)HIDC * HIDC * 2);
  __bf16* wpb  = (__bf16*)alloc((size_t)HIDC * HIDC * 2);
  __bf16* w1b  = (__bf16*)alloc((size_t)HIDC * MLPC * 2);
  __bf16* w2b  = (__bf16*)alloc((size_t)MLPC * HIDC * 2);
  float*  qf   = (float*) alloc((size_t)Md * HIDC * 4);
  __bf16* kb16 = (__bf16*)alloc((size_t)Ms * HIDC * 2);
  __bf16* vb16 = (__bf16*)alloc((size_t)Ms * HIDC * 2);
  float*  xr   = (float*) alloc((size_t)Md * HIDC * 4);
  __bf16* eemb = (__bf16*)alloc((size_t)E * HIDC * 2);
  float*  alpha= (float*) alloc((size_t)E * HEADSC * 4);
  unsigned* amax=(unsigned*)alloc((size_t)Md * HEADSC * 4);
  float*  denom= (float*) alloc((size_t)Md * HEADSC * 4);
  float*  oacc = (float*) alloc((size_t)Md * HIDC * 4);
  __bf16* tb   = (__bf16*)alloc((size_t)Md * HIDC * 2);
  float*  outf = (float*) alloc((size_t)Md * HIDC * 4);
  __bf16* hb   = (__bf16*)alloc((size_t)Md * HIDC * 2);
  __bf16* h1b  = (__bf16*)alloc((size_t)Md * MLPC * 2);

  auto grd = [](size_t n) { return dim3((unsigned)((n + 255) / 256)); };

  // weight conversions fp32 [K,N] -> bf16 transposed [N,K]
  dim3 tb32(32, 8);
  gt_cvt_bf16_t<<<dim3(HIDC/32, HIDC/32), tb32, 0, stream>>>(Wq, wqb, HIDC, HIDC);
  gt_cvt_bf16_t<<<dim3(HIDC/32, HIDC/32), tb32, 0, stream>>>(Wk, wkb, HIDC, HIDC);
  gt_cvt_bf16_t<<<dim3(HIDC/32, HIDC/32), tb32, 0, stream>>>(Wv, wvb, HIDC, HIDC);
  gt_cvt_bf16_t<<<dim3(HIDC/32, HIDC/32), tb32, 0, stream>>>(Ws, wsb, HIDC, HIDC);
  gt_cvt_bf16_t<<<dim3(HIDC/32, HIDC/32), tb32, 0, stream>>>(Wp, wpb, HIDC, HIDC);
  gt_cvt_bf16_t<<<dim3(MLPC/32, HIDC/32), tb32, 0, stream>>>(W1, w1b, HIDC, MLPC);
  gt_cvt_bf16_t<<<dim3(HIDC/32, MLPC/32), tb32, 0, stream>>>(W2, w2b, MLPC, HIDC);

  // LayerNorms -> bf16 GEMM operands
  gt_ln_bf16<<<Ms, 128, 0, stream>>>(x_src, ln1g, ln1b, hs, Ms);
  gt_ln_bf16<<<Md, 128, 0, stream>>>(x_dst, ln2g, ln2b, hd, Md);

  // projections (WMMA GEMMs)
  gt_gemm_bf16<<<dim3(HIDC/128, Md/128), 256, 0, stream>>>(hd, wqb, bq, nullptr, qf, nullptr, Md, HIDC, HIDC, 0);
  gt_gemm_bf16<<<dim3(HIDC/128, Ms/128), 256, 0, stream>>>(hs, wkb, bk, nullptr, nullptr, kb16, Ms, HIDC, HIDC, 0);
  gt_gemm_bf16<<<dim3(HIDC/128, Ms/128), 256, 0, stream>>>(hs, wvb, bv, nullptr, nullptr, vb16, Ms, HIDC, HIDC, 0);
  gt_gemm_bf16<<<dim3(HIDC/128, Md/128), 256, 0, stream>>>(hd, wsb, bs, nullptr, xr, nullptr, Md, HIDC, HIDC, 0);

  // edge embedding
  gt_edge_embed<<<E, 128, 0, stream>>>(eab, trn, We, be, eemb, E);

  // init accumulators (harness poisons ws -> must reset every call)
  gt_fill_u32<<<grd((size_t)Md*HEADSC), 256, 0, stream>>>(amax, ENC_NEG_INF, (size_t)Md*HEADSC);
  gt_fill_u32<<<grd((size_t)Md*HEADSC), 256, 0, stream>>>((unsigned*)denom, 0u, (size_t)Md*HEADSC);
  gt_fill_u32<<<grd((size_t)Md*HIDC),   256, 0, stream>>>((unsigned*)oacc,  0u, (size_t)Md*HIDC);

  // segment softmax + weighted aggregation
  dim3 eg = grd((size_t)E * HEADSC);
  gt_alpha <<<eg, 256, 0, stream>>>(qf, kb16, eemb, row_s, row_d, alpha, amax, E);
  gt_expsum<<<eg, 256, 0, stream>>>(amax, row_d, alpha, denom, E);
  gt_msg   <<<eg, 256, 0, stream>>>(vb16, eemb, alpha, denom, row_s, row_d, oacc, E);

  // out = (agg + x_r) @ Wp + bp + x_dst
  gt_add_cvt_bf16<<<grd((size_t)Md*HIDC), 256, 0, stream>>>(oacc, xr, tb, (size_t)Md*HIDC);
  gt_gemm_bf16<<<dim3(HIDC/128, Md/128), 256, 0, stream>>>(tb, wpb, bp, x_dst, outf, nullptr, Md, HIDC, HIDC, 0);

  // MLP: gelu(ln(out) @ W1 + b1) @ W2 + b2 + out
  gt_ln_bf16<<<Md, 128, 0, stream>>>(outf, mlng, mlnb, hb, Md);
  gt_gemm_bf16<<<dim3(MLPC/128, Md/128), 256, 0, stream>>>(hb, w1b, b1, nullptr, nullptr, h1b, Md, MLPC, HIDC, 1);
  gt_gemm_bf16<<<dim3(HIDC/128, Md/128), 256, 0, stream>>>(h1b, w2b, b2, outf, (float*)d_out, nullptr, Md, HIDC, MLPC, 0);
}
